// RelativeMultiheadAttention_46420006535353
// MI455X (gfx1250) — matrix operand order repaired and checked by reading, not compiled, not run on hardware
//
#include <hip/hip_runtime.h>

#define D_MODEL 1024
#define NUM_HEADS 16
#define D_K 64
#define MAX_REL 64
#define NREL (2 * MAX_REL - 1)   // 127
#define SEQ 2048
#define BATCH 2
#define ROWS (BATCH * SEQ)       // 4096

typedef __attribute__((ext_vector_type(16))) _Float16 v16h;
typedef __attribute__((ext_vector_type(4)))  _Float16 v4h;
typedef __attribute__((ext_vector_type(8)))  float    v8f;
typedef int vi4 __attribute__((vector_size(16)));

union AccF { v8f v; float f[8]; };
union Frag { v16h v; uint4 q[2]; };

#if __has_builtin(__builtin_amdgcn_global_load_async_to_lds_b128)
#define HAVE_ASYNC_LDS 1
#else
#define HAVE_ASYNC_LDS 0
#endif

#define AS1 __attribute__((address_space(1)))
#define AS3 __attribute__((address_space(3)))

// 16-byte global->LDS copy: async (ASYNCcnt-tracked, no VGPR staging) when available.
__device__ __forceinline__ void copy16_g2l(const _Float16* g, _Float16* l) {
#if HAVE_ASYNC_LDS
  __builtin_amdgcn_global_load_async_to_lds_b128((AS1 vi4*)g, (AS3 vi4*)l, 0, 0);
#else
  *(uint4*)l = *(const uint4*)g;
#endif
}

__device__ __forceinline__ void copy_wait() {
#if HAVE_ASYNC_LDS
#if __has_builtin(__builtin_amdgcn_s_wait_asynccnt)
  __builtin_amdgcn_s_wait_asynccnt(0);
#else
  asm volatile("s_wait_asynccnt 0" ::: "memory");
#endif
#endif
}

// DPP16 lane permutes (pure VALU, no LDS traffic; legal because values become
// progressively quad-/octet-uniform during the butterfly).
template <int CTRL>
__device__ __forceinline__ float dpp_f(float v) {
  int i = __builtin_bit_cast(int, v);
  i = __builtin_amdgcn_update_dpp(0, i, CTRL, 0xf, 0xf, true);
  return __builtin_bit_cast(float, i);
}
#define DPP_XOR1 0xB1   // quad_perm(1,0,3,2)
#define DPP_XOR2 0x4E   // quad_perm(2,3,0,1)
#define DPP_HMIR 0x141  // row_half_mirror
#define DPP_MIR  0x140  // row_mirror

__device__ __forceinline__ float redmax16(float x) {
  x = fmaxf(x, dpp_f<DPP_XOR1>(x));
  x = fmaxf(x, dpp_f<DPP_XOR2>(x));
  x = fmaxf(x, dpp_f<DPP_HMIR>(x));
  x = fmaxf(x, dpp_f<DPP_MIR>(x));
  return x;
}
__device__ __forceinline__ float redsum16(float x) {
  x += dpp_f<DPP_XOR1>(x);
  x += dpp_f<DPP_XOR2>(x);
  x += dpp_f<DPP_HMIR>(x);
  x += dpp_f<DPP_MIR>(x);
  return x;
}

// A-matrix fragment (16x32 f16, row-major source, stride ld in elements).
// Lane l: row m = l&15, half = l>>4. VGPR0-3: K = 8*half + 0..7 ; VGPR4-7: K = 16 + 8*half + 0..7.
__device__ __forceinline__ v16h ldA(const _Float16* base, int ld) {
  const int lane = threadIdx.x & 31;
  const int m = lane & 15;
  const int h8 = (lane >> 4) * 8;
  const _Float16* r = base + (size_t)m * ld;
  Frag f;
  f.q[0] = *(const uint4*)(r + h8);
  f.q[1] = *(const uint4*)(r + 16 + h8);
  return f.v;
}

// B-matrix fragment (32x16 f16) loaded from B^T stored row-major (rows = N, stride ld).
// Lane l: col n = l&15, half = l>>4. VGPR0-3: K = 8*half + 0..7 ; VGPR4-7: K = 16 + 8*half + 0..7.
__device__ __forceinline__ v16h ldBT(const _Float16* base, int ld) {
  const int lane = threadIdx.x & 31;
  const int n = lane & 15;
  const int h16 = (lane >> 4) * 8;
  const _Float16* r = base + (size_t)n * ld + h16;
  Frag f;
  f.q[0] = *(const uint4*)(r);
  f.q[1] = *(const uint4*)(r + 16);
  return f.v;
}

__device__ __forceinline__ v8f wmma_f16(v16h a, v16h b, v8f c) {
  return __builtin_amdgcn_wmma_f32_16x16x32_f16(false, a, false, b, (short)0, c, false, false);
}

// ---------------- prep kernels ----------------

__global__ void cvt_f32_f16(const float* __restrict__ in, _Float16* __restrict__ out, int n4) {
  int i = blockIdx.x * blockDim.x + threadIdx.x;
  if (i < n4) {
    float4 v = ((const float4*)in)[i];
    v4h h;
    h[0] = (_Float16)v.x; h[1] = (_Float16)v.y; h[2] = (_Float16)v.z; h[3] = (_Float16)v.w;
    ((v4h*)out)[i] = h;
  }
}

// Wt[n][k] = (f16) W[k][n], both 1024x1024
__global__ void wt_kernel(const float* __restrict__ W, _Float16* __restrict__ Wt) {
  __shared__ float t[32][33];
  const int bx = blockIdx.x, by = blockIdx.y;
  const int tx = threadIdx.x;
  for (int i = threadIdx.y; i < 32; i += 8)
    t[i][tx] = W[(size_t)(by * 32 + i) * D_MODEL + bx * 32 + tx];
  __syncthreads();
  for (int i = threadIdx.y; i < 32; i += 8)
    Wt[(size_t)(bx * 32 + i) * D_MODEL + by * 32 + tx] = (_Float16)t[tx][i];
}

// ---------------- projection GEMM ----------------
// C[4096x1024] = X[4096x1024] @ Wt^T + bias. BM=BN=128, BK=32, 256 threads (8 waves, 2x4).
// MODE 0: f16 head-major Q/K [bh][pos][64], (val+bias)*scale
// MODE 1: f16 transposed V   [bh][64][pos]
// MODE 2: f32 to d_out       [m][n]
template <int MODE>
__global__ void __launch_bounds__(256)
proj_gemm(const _Float16* __restrict__ X, const _Float16* __restrict__ Wt,
          const float* __restrict__ bias, void* __restrict__ outp, float scale) {
  __shared__ alignas(16) _Float16 As[128 * 40];
  __shared__ alignas(16) _Float16 Bs[128 * 40];
  const int tid = threadIdx.x;
  const int lane = tid & 31, wave = tid >> 5;
  const int wm = wave & 1, wn = wave >> 1;
  const int bm = blockIdx.x, bn = blockIdx.y;

  AccF acc[4][2];
#pragma unroll
  for (int mt = 0; mt < 4; mt++)
#pragma unroll
    for (int nt = 0; nt < 2; nt++)
#pragma unroll
      for (int j = 0; j < 8; j++) acc[mt][nt].f[j] = 0.f;

  for (int kk = 0; kk < D_MODEL; kk += 32) {
#pragma unroll
    for (int i = 0; i < 2; i++) {
      int slot = tid + 256 * i;
      int r = slot >> 2, c = (slot & 3) * 8;
      copy16_g2l(&X[(size_t)(bm * 128 + r) * D_MODEL + kk + c], &As[r * 40 + c]);
      copy16_g2l(&Wt[(size_t)(bn * 128 + r) * D_MODEL + kk + c], &Bs[r * 40 + c]);
    }
    copy_wait();
    __syncthreads();
    v16h fa[4], fb[2];
#pragma unroll
    for (int mt = 0; mt < 4; mt++) fa[mt] = ldA(&As[(wm * 64 + mt * 16) * 40], 40);
#pragma unroll
    for (int nt = 0; nt < 2; nt++) fb[nt] = ldBT(&Bs[(wn * 32 + nt * 16) * 40], 40);
#pragma unroll
    for (int mt = 0; mt < 4; mt++)
#pragma unroll
      for (int nt = 0; nt < 2; nt++)
        acc[mt][nt].v = wmma_f16(fa[mt], fb[nt], acc[mt][nt].v);
    __syncthreads();
  }

  const int nl = lane & 15, half = lane >> 4;
#pragma unroll
  for (int mt = 0; mt < 4; mt++)
#pragma unroll
    for (int nt = 0; nt < 2; nt++)
#pragma unroll
      for (int j = 0; j < 8; j++) {
        int m = bm * 128 + wm * 64 + mt * 16 + j + 8 * half;
        int n = bn * 128 + wn * 32 + nt * 16 + nl;
        float val = acc[mt][nt].f[j] + bias[n];
        if (MODE == 0) {
          int b = m >> 11, pos = m & 2047, h = n >> 6, d = n & 63;
          ((_Float16*)outp)[((size_t)((b * NUM_HEADS + h) * SEQ + pos)) * D_K + d] =
              (_Float16)(val * scale);
        } else if (MODE == 1) {
          int b = m >> 11, pos = m & 2047, h = n >> 6, d = n & 63;
          ((_Float16*)outp)[((size_t)((b * NUM_HEADS + h) * D_K + d)) * SEQ + pos] =
              (_Float16)val;
        } else {
          ((float*)outp)[(size_t)m * D_MODEL + n] = val;
        }
      }
}

// ---------------- flash attention ----------------
// Block: 128 threads = 4 waves; each wave owns one 16-row q-tile; all share (b,h).
// K/V blocks (64x64 f16 each) staged to LDS, double-buffered with async copies.
__global__ void __launch_bounds__(128)
attn_kernel(const _Float16* __restrict__ Qh, const _Float16* __restrict__ Kh,
            const _Float16* __restrict__ Vt, const float* __restrict__ rel_bias,
            const unsigned char* __restrict__ mask, _Float16* __restrict__ O16) {
  __shared__ alignas(16) _Float16 Ks[2][64 * 72];  // rows = kpos, cols = d (stride 72)
  __shared__ alignas(16) _Float16 Vs[2][64 * 72];  // rows = d,    cols = kpos
  __shared__ alignas(16) _Float16 Ps[4][16 * 72];  // per-wave P tile
  __shared__ float biasS[NREL + 1];
  const int bh = blockIdx.y, h = bh & (NUM_HEADS - 1), b = bh >> 4;
  const int wave = threadIdx.x >> 5, lane = threadIdx.x & 31;
  const int nl = lane & 15, half = lane >> 4;

  if (threadIdx.x < NREL) biasS[threadIdx.x] = rel_bias[h * NREL + threadIdx.x];

  const int qbase = (blockIdx.x * 4 + wave) * 16;
  const _Float16* Qb = Qh + (size_t)bh * SEQ * D_K;
  const _Float16* Kb = Kh + (size_t)bh * SEQ * D_K;
  const _Float16* Vb = Vt + (size_t)bh * D_K * SEQ;
  const unsigned char* mb = mask + (size_t)b * SEQ;

  // cooperative stage of one 64-key block of K and V into LDS buffer `buf`
  auto stage = [&](int buf, int kb2) {
#pragma unroll
    for (int i = 0; i < 4; i++) {
      int id = threadIdx.x + 128 * i;   // 512 chunks of 16B
      int r = id >> 3, c = (id & 7) * 8;
      copy16_g2l(Kb + (size_t)(kb2 + r) * D_K + c, &Ks[buf][r * 72 + c]);
      copy16_g2l(Vb + (size_t)r * SEQ + kb2 + c, &Vs[buf][r * 72 + c]);
    }
  };

  v16h fq[2];
  fq[0] = ldA(Qb + (size_t)qbase * D_K, D_K);
  fq[1] = ldA(Qb + (size_t)qbase * D_K + 32, D_K);

  AccF o[4];
  float rmax[8], rsum[8];
#pragma unroll
  for (int t = 0; t < 4; t++)
#pragma unroll
    for (int j = 0; j < 8; j++) o[t].f[j] = 0.f;
#pragma unroll
  for (int j = 0; j < 8; j++) { rmax[j] = -__builtin_inff(); rsum[j] = 0.f; }

  _Float16* Pw = &Ps[wave][0];

  stage(0, 0);

  for (int kb = 0; kb < SEQ; kb += 64) {
    const int buf = (kb >> 6) & 1;
    copy_wait();          // this wave's async copies (incl. next-buf issued last iter)
    __syncthreads();      // all waves' copies for `buf` visible; prev reads of buf^1 done
    if (kb + 64 < SEQ) stage(buf ^ 1, kb + 64);  // overlap next block's DMA with compute

    // S = Q @ K^T (1/sqrt(dk) folded into Q)
    AccF s[4];
#pragma unroll
    for (int t = 0; t < 4; t++) {
#pragma unroll
      for (int j = 0; j < 8; j++) s[t].f[j] = 0.f;
#pragma unroll
      for (int c = 0; c < 2; c++) {
        v16h fk = ldBT(&Ks[buf][(t * 16) * 72 + c * 32], 72);
        s[t].v = wmma_f16(fq[c], fk, s[t].v);
      }
    }

    // rel-bias: wave-uniform fast path when the whole 64-key block is clamped
    const bool uniLo = (kb >= qbase + 78);     // all rel <= -63 -> index 0
    const bool uniHi = (kb + 126 <= qbase);    // all rel >= +63 -> index 126
#pragma unroll
    for (int t = 0; t < 4; t++) {
      int kpos = kb + t * 16 + nl;
      bool mk = mb[kpos] != 0;
      if (uniLo || uniHi) {
        float bc = uniLo ? biasS[0] : biasS[NREL - 1];
#pragma unroll
        for (int j = 0; j < 8; j++) {
          float sv = s[t].f[j] + bc;
          s[t].f[j] = mk ? -1.0e30f : sv;
        }
      } else {
#pragma unroll
        for (int j = 0; j < 8; j++) {
          int q = qbase + j + 8 * half;
          int rel = q - kpos;
          rel = rel < -(MAX_REL - 1) ? -(MAX_REL - 1) : (rel > MAX_REL - 1 ? MAX_REL - 1 : rel);
          float sv = s[t].f[j] + biasS[rel + MAX_REL - 1];
          s[t].f[j] = mk ? -1.0e30f : sv;
        }
      }
    }

    // online softmax: DPP16 butterfly reductions within 16-lane halves (no LDS traffic)
    float newm[8], scl[8];
#pragma unroll
    for (int j = 0; j < 8; j++) {
      float mx = fmaxf(fmaxf(s[0].f[j], s[1].f[j]), fmaxf(s[2].f[j], s[3].f[j]));
      mx = redmax16(mx);
      newm[j] = fmaxf(rmax[j], mx);
      scl[j] = __expf(rmax[j] - newm[j]);
      rmax[j] = newm[j];
    }
    float bs[8];
#pragma unroll
    for (int j = 0; j < 8; j++) bs[j] = 0.f;
#pragma unroll
    for (int t = 0; t < 4; t++)
#pragma unroll
      for (int j = 0; j < 8; j++) {
        float p = __expf(s[t].f[j] - newm[j]);
        s[t].f[j] = p;
        bs[j] += p;
      }
#pragma unroll
    for (int j = 0; j < 8; j++) {
      rsum[j] = rsum[j] * scl[j] + redsum16(bs[j]);
    }
    // rescale O, spill P (C-layout -> LDS -> A-layout)
#pragma unroll
    for (int t = 0; t < 4; t++)
#pragma unroll
      for (int j = 0; j < 8; j++) {
        o[t].f[j] *= scl[j];
        Pw[(j + 8 * half) * 72 + t * 16 + nl] = (_Float16)s[t].f[j];
      }
    // O += P @ V
#pragma unroll
    for (int c = 0; c < 2; c++) {
      v16h fp = ldA(Pw + c * 32, 72);
#pragma unroll
      for (int t = 0; t < 4; t++) {
        v16h fv = ldBT(&Vs[buf][(t * 16) * 72 + c * 32], 72);
        o[t].v = wmma_f16(fp, fv, o[t].v);
      }
    }
    __syncthreads();  // everyone done reading `buf` before it is restaged
  }
  // normalize + write model-layout f16 for the output projection
#pragma unroll
  for (int t = 0; t < 4; t++)
#pragma unroll
    for (int j = 0; j < 8; j++) {
      int m = qbase + j + 8 * half;
      int col = h * D_K + t * 16 + nl;
      O16[(size_t)(b * SEQ + m) * D_MODEL + col] = (_Float16)(o[t].f[j] / rsum[j]);
    }
}

// ---------------- host ----------------

extern "C" void kernel_launch(void* const* d_in, const int* in_sizes, int n_in,
                              void* d_out, int out_size, void* d_ws, size_t ws_size,
                              hipStream_t stream) {
  (void)in_sizes; (void)n_in; (void)out_size; (void)ws_size;
  const float* query = (const float*)d_in[0];
  const float* key   = (const float*)d_in[1];
  const float* value = (const float*)d_in[2];
  const unsigned char* kpm = (const unsigned char*)d_in[3];
  const float* Wq = (const float*)d_in[4];
  const float* bq = (const float*)d_in[5];
  const float* Wk = (const float*)d_in[6];
  const float* bk = (const float*)d_in[7];
  const float* Wv = (const float*)d_in[8];
  const float* bv = (const float*)d_in[9];
  const float* Wo = (const float*)d_in[10];
  const float* bo = (const float*)d_in[11];
  const float* rb = (const float*)d_in[12];

  char* ws = (char*)d_ws;
  const size_t MB = 1024 * 1024;
  _Float16* Xq  = (_Float16*)(ws + 0 * MB);   // 8MB (reused as O16 after Q proj)
  _Float16* Xk  = (_Float16*)(ws + 8 * MB);
  _Float16* Xv  = (_Float16*)(ws + 16 * MB);
  _Float16* Wqt = (_Float16*)(ws + 24 * MB);
  _Float16* Wkt = (_Float16*)(ws + 26 * MB);
  _Float16* Wvt = (_Float16*)(ws + 28 * MB);
  _Float16* Wot = (_Float16*)(ws + 30 * MB);
  _Float16* Qh  = (_Float16*)(ws + 32 * MB);  // [32][2048][64]
  _Float16* Kh  = (_Float16*)(ws + 40 * MB);  // [32][2048][64]
  _Float16* Vt  = (_Float16*)(ws + 48 * MB);  // [32][64][2048]
  _Float16* O16 = Xq;                         // [4096][1024]

  int n4 = ROWS * D_MODEL / 4;
  dim3 cb(256), cg((n4 + 255) / 256);
  cvt_f32_f16<<<cg, cb, 0, stream>>>(query, Xq, n4);
  cvt_f32_f16<<<cg, cb, 0, stream>>>(key,   Xk, n4);
  cvt_f32_f16<<<cg, cb, 0, stream>>>(value, Xv, n4);

  dim3 tb(32, 8), tg(D_MODEL / 32, D_MODEL / 32);
  wt_kernel<<<tg, tb, 0, stream>>>(Wq, Wqt);
  wt_kernel<<<tg, tb, 0, stream>>>(Wk, Wkt);
  wt_kernel<<<tg, tb, 0, stream>>>(Wv, Wvt);
  wt_kernel<<<tg, tb, 0, stream>>>(Wo, Wot);

  dim3 gb(256), gg(ROWS / 128, D_MODEL / 128);
  proj_gemm<0><<<gg, gb, 0, stream>>>(Xq, Wqt, bq, (void*)Qh, 0.125f);  // 1/sqrt(64) folded
  proj_gemm<0><<<gg, gb, 0, stream>>>(Xk, Wkt, bk, (void*)Kh, 1.0f);
  proj_gemm<1><<<gg, gb, 0, stream>>>(Xv, Wvt, bv, (void*)Vt, 1.0f);

  dim3 ab(128), ag(SEQ / 64, BATCH * NUM_HEADS);
  attn_kernel<<<ag, ab, 0, stream>>>(Qh, Kh, Vt, rb, kpm, O16);

  proj_gemm<2><<<gg, gb, 0, stream>>>(O16, Wot, bo, d_out, 1.0f);
}
